// SetTransformer_ISAB_PMA_SAB_43885975830888
// MI455X (gfx1250) — compile-verified, matrix operand order checked by reading, not run on hardware
//
#include <hip/hip_runtime.h>
#include <hip/hip_bf16.h>
#include <math.h>
#include <stdint.h>

// Set Transformer (ISAB->ISAB->PMA->SAB->SAB->Linear) forward on gfx1250.
// All GEMMs and attention matmuls go through v_wmma_f32_16x16x32_bf16
// (fp32 inputs converted once to bf16; fp32 accumulation throughout).

#define ST_B     64
#define ST_N     1024
#define ST_DIN   256
#define ST_DH    512
#define ST_HEADS 8
#define ST_DHEAD 64
#define ST_NINDS 64

typedef __attribute__((ext_vector_type(16))) __bf16 v16bf;
typedef __attribute__((ext_vector_type(8)))  float  v8f;

union FragU { v16bf v; uint4 u[2]; __bf16 e[16]; };

// A/B fragment for wmma_f32_16x16x32_bf16 (wave32).
// Per ISA 16-bit A layout: lane L holds row (L&15); elements 0..7 = K (L>>4)*8+{0..7},
// elements 8..15 = K (L>>4)*8+16+{0..7}. Caller passes p = &M[row][k0 + (lane>>4)*8].
__device__ inline v16bf ld_frag(const __bf16* p) {
  FragU f;
  f.u[0] = *reinterpret_cast<const uint4*>(p);
  f.u[1] = *reinterpret_cast<const uint4*>(p + 16);
  return f.v;
}
__device__ inline v16bf zero_frag() { FragU f{}; return f.v; }

__device__ inline v8f wmma_bf16(v16bf a, v16bf b, v8f c) {
  return __builtin_amdgcn_wmma_f32_16x16x32_bf16(false, a, false, b, (short)0, c, false, false);
}

// ---------------------------------------------------------------------------
// fp32 -> bf16 convert with broadcast (dst[i] = src[i % srcN])
// ---------------------------------------------------------------------------
__global__ __launch_bounds__(256)
void cvt_bcast_kernel(const float* __restrict__ src, __bf16* __restrict__ dst,
                      int total, int srcN) {
  int i = blockIdx.x * 256 + threadIdx.x;
  if (i < total) dst[i] = (__bf16)src[i % srcN];
}

// ---------------------------------------------------------------------------
// Y = X @ W^T + bias   (X:[rows,din] bf16, W:[dout,din] bf16, row-major)
//   relu_res==0: Y stored (optionally) fp32 to Yf and bf16 to Yb
//   relu_res==1: Y = Res + max(0, X@W^T + bias)   (Res fp32, stride dout)
// rows % 64 == 0, dout % 64 == 0, din % 32 == 0 (all true for this model).
// Block = 128 threads (4 waves); block tile 64x64; wave tile 16x64.
// ---------------------------------------------------------------------------
__global__ __launch_bounds__(128)
void gemm_wmma_kernel(const __bf16* __restrict__ X, const __bf16* __restrict__ W,
                      const float* __restrict__ bias, const float* __restrict__ Res,
                      float* __restrict__ Yf, __bf16* __restrict__ Yb,
                      int din, int dout, int relu_res)
{
  const int lane = threadIdx.x & 31;
  const int wave = threadIdx.x >> 5;
  const int m0   = blockIdx.x * 64 + wave * 16;
  const int n0   = blockIdx.y * 64;
  const int lm   = lane & 15;
  const int kb8  = (lane >> 4) * 8;

  v8f acc0 = {}, acc1 = {}, acc2 = {}, acc3 = {};
  v8f* acc[4] = {&acc0, &acc1, &acc2, &acc3};

  const __bf16* xrow = X + (size_t)(m0 + lm) * din + kb8;
  for (int k0 = 0; k0 < din; k0 += 32) {
    v16bf a = ld_frag(xrow + k0);
#pragma unroll
    for (int t = 0; t < 4; t++) {
      // B[k,n] = W[n,k] -> lane n reads row (n0+t*16+lm) of W, contiguous in k.
      v16bf b = ld_frag(W + (size_t)(n0 + t * 16 + lm) * din + k0 + kb8);
      *acc[t] = wmma_bf16(a, b, *acc[t]);
    }
  }

  const int rbase = m0 + (lane >> 4) * 8;   // D element i -> row rbase+i, col tile+lm
#pragma unroll
  for (int t = 0; t < 4; t++) {
    const int c = n0 + t * 16 + lm;
    const float bv = bias[c];
#pragma unroll
    for (int i = 0; i < 8; i++) {
      const size_t idx = (size_t)(rbase + i) * dout + c;
      float v = (*acc[t])[i] + bv;
      if (relu_res) v = Res[idx] + fmaxf(v, 0.0f);
      if (Yf) Yf[idx] = v;
      if (Yb) Yb[idx] = (__bf16)v;
    }
  }
}

// ---------------------------------------------------------------------------
// Fused multi-head attention with residual:  of = q + softmax(q k^T * scale) v
// One wave32 per (batch b, head h, 16-row q tile). Flash-style online softmax,
// all state in registers; P staged via 1KB LDS tile to feed the WMMA A layout.
// q/k/v: [B*n, 512] (bf16 for matmuls, qf fp32 for residual). d_head = 64.
// ---------------------------------------------------------------------------
__global__ __launch_bounds__(32)
void attn_wmma_kernel(const float* __restrict__ qf, const __bf16* __restrict__ qb,
                      const __bf16* __restrict__ kbuf, const __bf16* __restrict__ vbuf,
                      float* __restrict__ of, __bf16* __restrict__ ob,
                      int nq, int nk, float scale)
{
  __shared__ __bf16 Plds[16][32];     // P tile, rows x 32 kv-cols

  const int lane = threadIdx.x & 31;
  const int lm   = lane & 15;
  const int hi   = lane >> 4;
  const int kb8  = hi * 8;
  const int qt   = blockIdx.x * 16;
  const int h    = blockIdx.y;
  const int b    = blockIdx.z;
  const int hoff = h * ST_DHEAD;

  // Q fragments covering d = 0..31 and 32..63 (rows padded with zeros)
  v16bf qa0, qa1;
  {
    const int qrl = qt + lm;
    if (qrl < nq) {
      const __bf16* qp = qb + ((size_t)(b * nq + qrl) * ST_DH + hoff + kb8);
      qa0 = ld_frag(qp);
      qa1 = ld_frag(qp + 32);
    } else { qa0 = zero_frag(); qa1 = zero_frag(); }
  }

  float M[8], L[8];
  v8f O0 = {}, O1v = {}, O2v = {}, O3v = {};
  v8f* O[4] = {&O0, &O1v, &O2v, &O3v};
#pragma unroll
  for (int i = 0; i < 8; i++) { M[i] = -1e30f; L[i] = 0.0f; }

  for (int j = 0; j < nk; j += 32) {
    // ---- S = scale * q k^T : two 16x16 tiles (cols j..j+15, j+16..j+31) ----
    v8f sA = {}, sB = {};
    {
      int kr = j + lm;
      v16bf bk0, bk1;
      if (kr < nk) {
        const __bf16* kp = kbuf + ((size_t)(b * nk + kr) * ST_DH + hoff + kb8);
        bk0 = ld_frag(kp); bk1 = ld_frag(kp + 32);
      } else { bk0 = zero_frag(); bk1 = zero_frag(); }
      sA = wmma_bf16(qa0, bk0, sA);
      sA = wmma_bf16(qa1, bk1, sA);
      kr = j + 16 + lm;
      if (kr < nk) {
        const __bf16* kp = kbuf + ((size_t)(b * nk + kr) * ST_DH + hoff + kb8);
        bk0 = ld_frag(kp); bk1 = ld_frag(kp + 32);
      } else { bk0 = zero_frag(); bk1 = zero_frag(); }
      sB = wmma_bf16(qa0, bk0, sB);
      sB = wmma_bf16(qa1, bk1, sB);
    }

    // ---- online softmax update (row stats via 16-lane xor shuffles) ----
    const int c0 = j + lm, c1 = j + 16 + lm;
#pragma unroll
    for (int i = 0; i < 8; i++) {
      float x0 = (c0 < nk) ? sA[i] * scale : -1e30f;
      float x1 = (c1 < nk) ? sB[i] * scale : -1e30f;
      float m = fmaxf(x0, x1);
#pragma unroll
      for (int d = 1; d < 16; d <<= 1) m = fmaxf(m, __shfl_xor(m, d, 32));
      const float mnew = fmaxf(M[i], m);
      const float corr = __expf(M[i] - mnew);
      const float p0 = __expf(x0 - mnew);
      const float p1 = __expf(x1 - mnew);
      float rs = p0 + p1;
#pragma unroll
      for (int d = 1; d < 16; d <<= 1) rs += __shfl_xor(rs, d, 32);
      L[i] = L[i] * corr + rs;
      M[i] = mnew;
      (*O[0])[i] *= corr; (*O[1])[i] *= corr;
      (*O[2])[i] *= corr; (*O[3])[i] *= corr;
      Plds[i + 8 * hi][lm]      = (__bf16)p0;
      Plds[i + 8 * hi][16 + lm] = (__bf16)p1;
    }
    __syncthreads();

    // ---- O += P @ V  (A = P from LDS; B = V tile, strided bf16 loads) ----
    v16bf pa = ld_frag(&Plds[lm][kb8]);
#pragma unroll
    for (int t = 0; t < 4; t++) {
      FragU f;
      const int col = hoff + t * 16 + lm;
#pragma unroll
      for (int e = 0; e < 16; e++) {
        const int kk = kb8 + e + ((e & 8) ? 8 : 0);
        const int vr = j + kk;
        f.e[e] = (vr < nk) ? vbuf[(size_t)(b * nk + vr) * ST_DH + col]
                           : (__bf16)0.0f;
      }
      *O[t] = wmma_bf16(pa, f.v, *O[t]);
    }
    __syncthreads();
  }

  // ---- epilogue: normalize, add q residual, store fp32 + bf16 ----
#pragma unroll
  for (int t = 0; t < 4; t++) {
    const int col = hoff + t * 16 + lm;
#pragma unroll
    for (int i = 0; i < 8; i++) {
      const int rl = qt + 8 * hi + i;
      if (rl < nq) {
        const size_t idx = (size_t)(b * nq + rl) * ST_DH + col;
        const float v = (*O[t])[i] / L[i] + qf[idx];
        of[idx] = v;
        ob[idx] = (__bf16)v;
      }
    }
  }
}

// ---------------------------------------------------------------------------
// Host orchestration
// ---------------------------------------------------------------------------
extern "C" void kernel_launch(void* const* d_in, const int* in_sizes, int n_in,
                              void* d_out, int out_size, void* d_ws, size_t ws_size,
                              hipStream_t stream)
{
  (void)in_sizes; (void)n_in; (void)out_size; (void)ws_size;
  // Input layout (setup_inputs dict order, params recursively in insertion order):
  // 0:x 1:I1 2..9:isab1_mab0{qW,qb,kW,kb,vW,vb,oW,ob} 10..17:isab1_mab1
  // 18:I2 19..26:isab2_mab0 27..34:isab2_mab1 35:S 36..43:pma 44..51:sab1
  // 52..59:sab2 60:fcW 61:fcb
  const float* x   = (const float*)d_in[0];
  const float* I1  = (const float*)d_in[1];
  const float* I2  = (const float*)d_in[18];
  const float* Sp  = (const float*)d_in[35];
  const float* fcW = (const float*)d_in[60];
  const float* fcb = (const float*)d_in[61];

  char* ws = (char*)d_ws;
  size_t off = 0;
  auto alloc = [&](size_t bytes) -> char* {
    off = (off + 255) & ~(size_t)255;
    char* p = ws + off; off += bytes; return p;
  };

  const size_t RMAX = (size_t)ST_B * ST_N;            // 65536 rows max
  __bf16* xb  = (__bf16*)alloc((size_t)ST_B * ST_N * ST_DIN * 2);
  __bf16* ib  = (__bf16*)alloc((size_t)ST_B * ST_NINDS * ST_DH * 2); // I1/I2/S bcast
  float*  qf  = (float *)alloc(RMAX * ST_DH * 4);
  __bf16* qb2 = (__bf16*)alloc(RMAX * ST_DH * 2);
  __bf16* kb2 = (__bf16*)alloc(RMAX * ST_DH * 2);
  __bf16* vb2 = (__bf16*)alloc(RMAX * ST_DH * 2);
  float*  ofb = (float *)alloc(RMAX * ST_DH * 4);
  __bf16* ob2 = (__bf16*)alloc(RMAX * ST_DH * 2);
  __bf16* Oa  = (__bf16*)alloc(RMAX * ST_DH * 2);     // MAB output ping
  __bf16* Ob  = (__bf16*)alloc(RMAX * ST_DH * 2);     // MAB output pong

  auto cvt = [&](const float* src, __bf16* dst, int total, int srcN) {
    cvt_bcast_kernel<<<dim3((total + 255) / 256), dim3(256), 0, stream>>>(src, dst, total, srcN);
  };

  static const int mab_base[7] = {2, 10, 19, 27, 36, 44, 52};
  static const int mab_dq[7] = {ST_DH, ST_DIN, ST_DH, ST_DH, ST_DH, ST_DH, ST_DH};
  static const int mab_dk[7] = {ST_DIN, ST_DH, ST_DH, ST_DH, ST_DH, ST_DH, ST_DH};
  __bf16 *wq[7], *wk[7], *wv[7], *wo[7];
  for (int m = 0; m < 7; m++) {
    const int dq = mab_dq[m], dk = mab_dk[m];
    wq[m] = (__bf16*)alloc((size_t)ST_DH * dq * 2);
    wk[m] = (__bf16*)alloc((size_t)ST_DH * dk * 2);
    wv[m] = (__bf16*)alloc((size_t)ST_DH * dk * 2);
    wo[m] = (__bf16*)alloc((size_t)ST_DH * ST_DH * 2);
    cvt((const float*)d_in[mab_base[m] + 0], wq[m], ST_DH * dq, ST_DH * dq);
    cvt((const float*)d_in[mab_base[m] + 2], wk[m], ST_DH * dk, ST_DH * dk);
    cvt((const float*)d_in[mab_base[m] + 4], wv[m], ST_DH * dk, ST_DH * dk);
    cvt((const float*)d_in[mab_base[m] + 6], wo[m], ST_DH * ST_DH, ST_DH * ST_DH);
  }
  __bf16* fcWb = (__bf16*)alloc((size_t)ST_DH * ST_DH * 2);
  cvt(fcW, fcWb, ST_DH * ST_DH, ST_DH * ST_DH);

  cvt(x, xb, ST_B * ST_N * ST_DIN, ST_B * ST_N * ST_DIN);

  auto gemm = [&](const __bf16* X, const __bf16* W, const float* bias, const float* Res,
                  float* Yf, __bf16* Yb, int rows, int din, int dout, int relu) {
    gemm_wmma_kernel<<<dim3(rows / 64, dout / 64), dim3(128), 0, stream>>>(
        X, W, bias, Res, Yf, Yb, din, dout, relu);
  };

  const float scale = 1.0f / sqrtf((float)ST_DH);

  auto run_mab = [&](const __bf16* Qb, int nq, int dq,
                     const __bf16* Kb, int nk, int dk, int m, __bf16* outb) {
    const float* bq = (const float*)d_in[mab_base[m] + 1];
    const float* bk = (const float*)d_in[mab_base[m] + 3];
    const float* bv = (const float*)d_in[mab_base[m] + 5];
    const float* bo = (const float*)d_in[mab_base[m] + 7];
    const int rq = ST_B * nq, rk = ST_B * nk;
    gemm(Qb, wq[m], bq, nullptr, qf, qb2, rq, dq, ST_DH, 0);       // q (f32+bf16)
    gemm(Kb, wk[m], bk, nullptr, nullptr, kb2, rk, dk, ST_DH, 0);  // k (bf16)
    gemm(Kb, wv[m], bv, nullptr, nullptr, vb2, rk, dk, ST_DH, 0);  // v (bf16)
    attn_wmma_kernel<<<dim3((nq + 15) / 16, ST_HEADS, ST_B), dim3(32), 0, stream>>>(
        qf, qb2, kb2, vb2, ofb, ob2, nq, nk, scale);               // o = q + A v
    gemm(ob2, wo[m], bo, ofb, nullptr, outb, rq, ST_DH, ST_DH, 1); // o + relu(oW+b)
  };

  // Encoder: ISAB1
  cvt(I1, ib, ST_B * ST_NINDS * ST_DH, ST_NINDS * ST_DH);
  run_mab(ib, ST_NINDS, ST_DH, xb, ST_N, ST_DIN, 0, Oa);   // H1
  run_mab(xb, ST_N, ST_DIN, Oa, ST_NINDS, ST_DH, 1, Ob);   // x1
  // Encoder: ISAB2
  cvt(I2, ib, ST_B * ST_NINDS * ST_DH, ST_NINDS * ST_DH);
  run_mab(ib, ST_NINDS, ST_DH, Ob, ST_N, ST_DH, 2, Oa);    // H2
  run_mab(Ob, ST_N, ST_DH, Oa, ST_NINDS, ST_DH, 3, Ob);    // x2
  // Decoder: PMA -> SAB -> SAB
  cvt(Sp, ib, ST_B * 1 * ST_DH, ST_DH);
  run_mab(ib, 1, ST_DH, Ob, ST_N, ST_DH, 4, Oa);           // PMA
  run_mab(Oa, 1, ST_DH, Oa, 1, ST_DH, 5, Ob);              // SAB1
  run_mab(Ob, 1, ST_DH, Ob, 1, ST_DH, 6, Oa);              // SAB2
  // fc_out -> d_out [B, 512]
  gemm(Oa, fcWb, fcb, nullptr, (float*)d_out, nullptr, ST_B, ST_DH, ST_DH, 0);
}
